// Qwen3MoeSparseMoeBlock_28613072126657
// MI455X (gfx1250) — matrix operand order repaired and checked by reading, not compiled, not run on hardware
//
#include <hip/hip_runtime.h>
#include <hip/hip_bf16.h>

// ---------------- problem constants (Qwen3 MoE block) ----------------
#define HDIM 2048   // hidden
#define EXP  64     // experts
#define IDIM 768    // intermediate
#define TOK  2048   // B*S tokens
#define TOPK 8

typedef __attribute__((ext_vector_type(16))) __bf16 v16bf;
typedef __attribute__((ext_vector_type(8)))  float  v8f;

// native f32 -> bf16 convert
__device__ __forceinline__ unsigned short f2bf(float f) {
    __bf16 b = (__bf16)f;
    return __builtin_bit_cast(unsigned short, b);
}
// two f32 -> one packed bf16x2 dword: force exactly one v_cvt_pk_bf16_f32
__device__ __forceinline__ unsigned pk2bf(float lo, float hi) {
    unsigned r;
    asm("v_cvt_pk_bf16_f32 %0, %1, %2" : "=v"(r) : "v"(lo), "v"(hi));
    return r;
}

union ABReg { v16bf v; uint4 q[2]; };

// ---------------------------------------------------------------------
// 0) zero combine-output region + expert counts
// ---------------------------------------------------------------------
__global__ void moe_zero(float* __restrict__ out, int* __restrict__ counts) {
    long i = (long)blockIdx.x * blockDim.x + threadIdx.x;
    long i4 = i * 4;
    if (i4 < (long)TOK * HDIM) {
        float4 z = make_float4(0.f, 0.f, 0.f, 0.f);
        *(float4*)&out[i4] = z;
    }
    if (i < EXP) counts[i] = 0;
}

// ---------------------------------------------------------------------
// 1) router: logits -> d_out tail, softmax+top8, scatter to expert lists
// ---------------------------------------------------------------------
__global__ void moe_router(const float* __restrict__ x,
                           const float* __restrict__ rw,
                           float* __restrict__ logits_out,
                           int* __restrict__ counts,
                           int* __restrict__ tlist,
                           float* __restrict__ wlist) {
    const int t   = blockIdx.x;
    const int tid = threadIdx.x;            // 0..63 == expert id

    __shared__ __align__(16) float sX[HDIM];
    __shared__ float sL[EXP];
    __shared__ float sP[EXP];

    for (int i = tid; i < HDIM / 4; i += 64)
        *(float4*)&sX[i * 4] = *(const float4*)&x[(long)t * HDIM + i * 4];
    __syncthreads();

    float acc = 0.f;
    #pragma unroll 8
    for (int h = 0; h < HDIM; ++h)
        acc = fmaf(sX[h], rw[h * EXP + tid], acc);

    logits_out[t * EXP + tid] = acc;        // second reference output
    sL[tid] = acc;
    __syncthreads();

    float mx = sL[0];
    for (int i = 1; i < EXP; ++i) mx = fmaxf(mx, sL[i]);
    sP[tid] = __expf(acc - mx);
    __syncthreads();

    if (tid == 0) {
        int   idx[TOPK];
        float w[TOPK];
        float tw = 0.f;
        for (int k = 0; k < TOPK; ++k) {
            int   bj = 0;
            float bv = sP[0];
            for (int j = 1; j < EXP; ++j)
                if (sP[j] > bv) { bv = sP[j]; bj = j; }
            idx[k] = bj; w[k] = bv; tw += bv;
            sP[bj] = -1.f;
        }
        float inv = 1.f / tw;   // softmax denominator cancels in norm_topk
        for (int k = 0; k < TOPK; ++k) {
            int e   = idx[k];
            int pos = atomicAdd(&counts[e], 1);
            tlist[e * TOK + pos] = t;
            wlist[e * TOK + pos] = w[k] * inv;
        }
    }
}

// ---------------------------------------------------------------------
// 2) exclusive prefix sum of counts -> compact slot offsets
// ---------------------------------------------------------------------
__global__ void moe_prefix(const int* __restrict__ counts, int* __restrict__ offsets) {
    if (threadIdx.x == 0) {
        int acc = 0;
        for (int e = 0; e < EXP; ++e) { offsets[e] = acc; acc += counts[e]; }
        offsets[EXP] = acc;
    }
}

// ---------------------------------------------------------------------
// 3) gate+up GEMM: h = silu(x@Wg) * (x@Wu) * route_w  (bf16 WMMA)
//    block = 256 thr = 8 waves; tile BM=64 x BN=64; K-step 32
//    wave(mi,ni) owns two 16x16 N-subtiles of BOTH gate and up
//    -> 4 v_wmma per K-step per wave off one A fragment
// ---------------------------------------------------------------------
__global__ void moe_gateup(const float* __restrict__ x,
                           const float* __restrict__ gate_w,
                           const float* __restrict__ up_w,
                           const int*   __restrict__ counts,
                           const int*   __restrict__ offsets,
                           const int*   __restrict__ tlist,
                           const float* __restrict__ wlist,
                           unsigned short* __restrict__ hbuf) {
    const int e   = blockIdx.x;
    const int cnt = counts[e];
    const int m_base = blockIdx.y * 64;
    if (m_base >= cnt) return;                 // uniform: whole block exits
    const int n0  = blockIdx.z * 64;
    const int tid  = threadIdx.x;
    const int lane = tid & 31;
    const int mi   = (tid >> 5) & 3;           // M sub-tile 0..3
    const int ni   = (tid >> 5) >> 2;          // N half 0..1 (32 wide)

    __shared__ __align__(16) unsigned short smem[6144];
    unsigned short* sA  = smem;                // 64 rows x 32 k
    unsigned short* sBg = smem + 2048;         // [64 cols][32 k] col-major
    unsigned short* sBu = smem + 4096;         // [64 cols][32 k] col-major
    unsigned short* sH  = smem + 2048;         // 64x64 h tile (epilogue reuse)
    __shared__ int   sTok[64];
    __shared__ float sW[64];

    if (tid < 64) {
        int r = m_base + tid;
        if (r < cnt) { sTok[tid] = tlist[e * TOK + r]; sW[tid] = wlist[e * TOK + r]; }
        else         { sTok[tid] = 0;                  sW[tid] = 0.f; }
    }
    __syncthreads();

    v8f cg0 = {}, cg1 = {}, cu0 = {}, cu1 = {};
    const long gbase = (long)e * HDIM * IDIM;

    for (int k0 = 0; k0 < HDIM; k0 += 32) {
        // ---- stage A: 64x32 f32 -> bf16, packed cvt pairs on k ----
        #pragma unroll
        for (int t2 = 0; t2 < 2; ++t2) {
            int q   = tid * 2 + t2;
            int row = q >> 3;
            int kq  = (q & 7) * 4;
            float4 v = *(const float4*)&x[(long)sTok[row] * HDIM + k0 + kq];
            uint2 p;
            p.x = pk2bf(v.x, v.y);
            p.y = pk2bf(v.z, v.w);
            *(uint2*)&sA[row * 32 + kq] = p;
        }
        // ---- stage B gate/up: two k-rows per thread so (k,k+1) pairs
        //      land in one packed cvt + one ds_store_b32 (col-major) ----
        {
            int kr2 = (tid >> 4) * 2;          // 0,2,..,30
            int nq  = (tid & 15) * 4;          // 0..60
            long boff = gbase + (long)(k0 + kr2) * IDIM + n0 + nq;
            float4 ga = *(const float4*)&gate_w[boff];
            float4 gb = *(const float4*)&gate_w[boff + IDIM];
            float4 ua = *(const float4*)&up_w[boff];
            float4 ub = *(const float4*)&up_w[boff + IDIM];
            *(unsigned*)&sBg[(nq + 0) * 32 + kr2] = pk2bf(ga.x, gb.x);
            *(unsigned*)&sBg[(nq + 1) * 32 + kr2] = pk2bf(ga.y, gb.y);
            *(unsigned*)&sBg[(nq + 2) * 32 + kr2] = pk2bf(ga.z, gb.z);
            *(unsigned*)&sBg[(nq + 3) * 32 + kr2] = pk2bf(ga.w, gb.w);
            *(unsigned*)&sBu[(nq + 0) * 32 + kr2] = pk2bf(ua.x, ub.x);
            *(unsigned*)&sBu[(nq + 1) * 32 + kr2] = pk2bf(ua.y, ub.y);
            *(unsigned*)&sBu[(nq + 2) * 32 + kr2] = pk2bf(ua.z, ub.z);
            *(unsigned*)&sBu[(nq + 3) * 32 + kr2] = pk2bf(ua.w, ub.w);
            if (k0 + 32 < HDIM) {              // stream next weight tile
                __builtin_prefetch(&gate_w[boff + 32L * IDIM], 0, 1);
                __builtin_prefetch(&up_w[boff + 32L * IDIM], 0, 1);
            }
        }
        __syncthreads();

        // ---- lane fragments in contiguous k order (k = 16*(lane/16) + s), A and B alike ----
        ABReg a, bg0, bg1, bu0, bu1;
        int am = mi * 16 + (lane & 15);
        int ka = (lane & 16) ? 16 : 0;         // lanes16-31: K 16-31
        a.q[0] = *(const uint4*)&sA[am * 32 + ka];
        a.q[1] = *(const uint4*)&sA[am * 32 + ka + 8];
        int bn = ni * 32 + (lane & 15);
        int kb = (lane & 16) ? 16 : 0;         // lanes16-31: K 16-31
        bg0.q[0] = *(const uint4*)&sBg[bn * 32 + kb];
        bg0.q[1] = *(const uint4*)&sBg[bn * 32 + kb + 8];
        bg1.q[0] = *(const uint4*)&sBg[(bn + 16) * 32 + kb];
        bg1.q[1] = *(const uint4*)&sBg[(bn + 16) * 32 + kb + 8];
        bu0.q[0] = *(const uint4*)&sBu[bn * 32 + kb];
        bu0.q[1] = *(const uint4*)&sBu[bn * 32 + kb + 8];
        bu1.q[0] = *(const uint4*)&sBu[(bn + 16) * 32 + kb];
        bu1.q[1] = *(const uint4*)&sBu[(bn + 16) * 32 + kb + 8];

        cg0 = __builtin_amdgcn_wmma_f32_16x16x32_bf16(false, a.v, false, bg0.v,
                                                      (short)0, cg0, false, false);
        cg1 = __builtin_amdgcn_wmma_f32_16x16x32_bf16(false, a.v, false, bg1.v,
                                                      (short)0, cg1, false, false);
        cu0 = __builtin_amdgcn_wmma_f32_16x16x32_bf16(false, a.v, false, bu0.v,
                                                      (short)0, cu0, false, false);
        cu1 = __builtin_amdgcn_wmma_f32_16x16x32_bf16(false, a.v, false, bu1.v,
                                                      (short)0, cu1, false, false);
        __syncthreads();
    }

    // ---- fused SwiGLU + routing weight -> 64x64 bf16 tile in LDS ----
    {
        int cn    = ni * 32 + (lane & 15);
        int mrow0 = mi * 16 + ((lane & 16) ? 8 : 0);
        #pragma unroll
        for (int r = 0; r < 8; ++r) {
            int   m = mrow0 + r;
            float w = sW[m];
            float g0 = cg0[r], u0 = cu0[r];
            float g1 = cg1[r], u1 = cu1[r];
            sH[m * 64 + cn]      = f2bf((g0 / (1.f + __expf(-g0))) * u0 * w);
            sH[m * 64 + cn + 16] = f2bf((g1 / (1.f + __expf(-g1))) * u1 * w);
        }
    }
    __syncthreads();

    // ---- coalesced bf16 store of h tile (512 uint4, 2 per thread) ----
    #pragma unroll
    for (int t2 = 0; t2 < 2; ++t2) {
        int q   = tid * 2 + t2;
        int row = q >> 3;
        int seg = (q & 7) * 8;
        if (m_base + row < cnt) {
            long slot = (long)offsets[e] + m_base + row;
            *(uint4*)&hbuf[slot * IDIM + n0 + seg] = *(const uint4*)&sH[row * 64 + seg];
        }
    }
}

// ---------------------------------------------------------------------
// 4) down GEMM + weighted scatter-add:  out[tok] += h @ Wd[e]
//    tile BM=64 x BN=64; K=768 step 32; 2 v_wmma per K-step per wave
// ---------------------------------------------------------------------
__global__ void moe_down(const unsigned short* __restrict__ hbuf,
                         const float* __restrict__ down_w,
                         const int*   __restrict__ counts,
                         const int*   __restrict__ offsets,
                         const int*   __restrict__ tlist,
                         float* __restrict__ out) {
    const int e   = blockIdx.x;
    const int cnt = counts[e];
    const int m_base = blockIdx.y * 64;
    if (m_base >= cnt) return;
    const int n0  = blockIdx.z * 64;
    const int tid  = threadIdx.x;
    const int lane = tid & 31;
    const int mi   = (tid >> 5) & 3;
    const int ni   = (tid >> 5) >> 2;

    __shared__ __align__(16) unsigned short sA[64 * 32];   // bf16 h tile
    __shared__ __align__(16) unsigned short sB[64 * 32];   // col-major
    __shared__ int sTok[64];

    if (tid < 64) {
        int r = m_base + tid;
        sTok[tid] = (r < cnt) ? tlist[e * TOK + r] : 0;
    }
    __syncthreads();

    v8f c0 = {}, c1 = {};
    const long dbase     = (long)e * IDIM * HDIM;
    const int  base_slot = offsets[e];

    for (int k0 = 0; k0 < IDIM; k0 += 32) {
        // ---- stage A (already bf16): 64x32, one uint4 per thread ----
        {
            int row = tid >> 2;
            int seg = (tid & 3) * 8;
            uint4 av = make_uint4(0u, 0u, 0u, 0u);
            if (m_base + row < cnt)
                av = *(const uint4*)&hbuf[(long)(base_slot + m_base + row) * IDIM + k0 + seg];
            *(uint4*)&sA[row * 32 + seg] = av;
        }
        // ---- stage B: two k-rows/thread, packed cvt -> ds_store_b32 ----
        {
            int kr2 = (tid >> 4) * 2;
            int nq  = (tid & 15) * 4;
            long boff = dbase + (long)(k0 + kr2) * HDIM + n0 + nq;
            float4 da = *(const float4*)&down_w[boff];
            float4 db = *(const float4*)&down_w[boff + HDIM];
            *(unsigned*)&sB[(nq + 0) * 32 + kr2] = pk2bf(da.x, db.x);
            *(unsigned*)&sB[(nq + 1) * 32 + kr2] = pk2bf(da.y, db.y);
            *(unsigned*)&sB[(nq + 2) * 32 + kr2] = pk2bf(da.z, db.z);
            *(unsigned*)&sB[(nq + 3) * 32 + kr2] = pk2bf(da.w, db.w);
            if (k0 + 32 < IDIM)
                __builtin_prefetch(&down_w[boff + 32L * HDIM], 0, 1);
        }
        __syncthreads();

        ABReg a, b0, b1;
        int am = mi * 16 + (lane & 15);
        int ka = (lane & 16) ? 16 : 0;
        a.q[0] = *(const uint4*)&sA[am * 32 + ka];
        a.q[1] = *(const uint4*)&sA[am * 32 + ka + 8];
        int bn = ni * 32 + (lane & 15);
        int kb = (lane & 16) ? 16 : 0;
        b0.q[0] = *(const uint4*)&sB[bn * 32 + kb];
        b0.q[1] = *(const uint4*)&sB[bn * 32 + kb + 8];
        b1.q[0] = *(const uint4*)&sB[(bn + 16) * 32 + kb];
        b1.q[1] = *(const uint4*)&sB[(bn + 16) * 32 + kb + 8];

        c0 = __builtin_amdgcn_wmma_f32_16x16x32_bf16(false, a.v, false, b0.v,
                                                     (short)0, c0, false, false);
        c1 = __builtin_amdgcn_wmma_f32_16x16x32_bf16(false, a.v, false, b1.v,
                                                     (short)0, c1, false, false);
        __syncthreads();
    }

    // ---- weighted combine: atomic scatter-add straight from accumulators ----
    {
        int cn    = ni * 32 + (lane & 15);
        int mrow0 = mi * 16 + ((lane & 16) ? 8 : 0);
        #pragma unroll
        for (int r = 0; r < 8; ++r) {
            int m = mrow0 + r;
            if (m_base + m < cnt) {
                float* orow = &out[(long)sTok[m] * HDIM + n0];
                atomicAdd(orow + cn,      c0[r]);
                atomicAdd(orow + cn + 16, c1[r]);
            }
        }
    }
}

// ---------------------------------------------------------------------
// launch
// ---------------------------------------------------------------------
extern "C" void kernel_launch(void* const* d_in, const int* in_sizes, int n_in,
                              void* d_out, int out_size, void* d_ws, size_t ws_size,
                              hipStream_t stream) {
    const float* x      = (const float*)d_in[0];   // [2,1024,2048]
    const float* rw     = (const float*)d_in[1];   // [2048,64]
    const float* gate_w = (const float*)d_in[2];   // [64,2048,768]
    const float* up_w   = (const float*)d_in[3];   // [64,2048,768]
    const float* down_w = (const float*)d_in[4];   // [64,768,2048]
    float* out = (float*)d_out;                    // [T*H] out ++ [T*E] logits

    char* ws = (char*)d_ws;
    int*   counts  = (int*)(ws);                                   // 64
    int*   offsets = (int*)(ws + 256);                             // 65
    int*   tlist   = (int*)(ws + 1024);                            // 64*2048
    float* wlist   = (float*)(ws + 1024 + (size_t)EXP * TOK * 4);  // 64*2048
    unsigned short* hbuf =
        (unsigned short*)(ws + 1024 + 2 * (size_t)EXP * TOK * 4);  // 16384*768 bf16

    // 0) zero output combine region + counts
    {
        long n4 = ((long)TOK * HDIM) / 4;
        int blocks = (int)((n4 + 255) / 256);
        moe_zero<<<blocks, 256, 0, stream>>>(out, counts);
    }
    // 1) router (also writes router_logits to the output tail)
    moe_router<<<TOK, 64, 0, stream>>>(x, rw, out + (long)TOK * HDIM,
                                       counts, tlist, wlist);
    // 2) slot offsets
    moe_prefix<<<1, 32, 0, stream>>>(counts, offsets);
    // 3) gate/up + SwiGLU (bf16 WMMA)
    moe_gateup<<<dim3(EXP, TOK / 64, IDIM / 64), 256, 0, stream>>>(
        x, gate_w, up_w, counts, offsets, tlist, wlist, hbuf);
    // 4) down + weighted combine (bf16 WMMA + f32 atomics)
    moe_down<<<dim3(EXP, TOK / 64, HDIM / 64), 256, 0, stream>>>(
        hbuf, down_w, counts, offsets, tlist, out);
}
